// LSTMAutoencoder_25177098289633
// MI455X (gfx1250) — matrix operand order repaired and checked by reading, not compiled, not run on hardware
//
#include <hip/hip_runtime.h>

#define SEQ_LEN 256
#define BATCH 4096

typedef __attribute__((ext_vector_type(16))) _Float16 v16h;
typedef __attribute__((ext_vector_type(8)))  _Float16 v8h;
typedef __attribute__((ext_vector_type(8)))  float v8f;
typedef __attribute__((ext_vector_type(4)))  float v4f;

__device__ __forceinline__ float frcp(float x) {
  return __builtin_amdgcn_rcpf(x);           // v_rcp_f32 (no IEEE div fixup)
}
__device__ __forceinline__ float fsig(float x) {
  // sigmoid(x) = 1/(1+exp(-x)); exp->inf => rcp->0, exp->0 => 1  (no clamps needed)
  return frcp(1.f + __expf(-x));
}
__device__ __forceinline__ float ftanh(float x) {
  // tanh(x) = 1 - 2/(exp(2x)+1); saturates to +/-1 at infinities, no clamps needed
  return 1.f - 2.f * frcp(1.f + __expf(2.f * x));
}
__device__ __forceinline__ v8f bcast8(float v) {
  v8f r;
#pragma unroll
  for (int i = 0; i < 8; ++i) r[i] = v;
  return r;
}
__device__ __forceinline__ v8f wmma16(v16h a, v16h b, v8f c) {
  // D = A(16x32 f16) x B(32x16 f16) + C(16x16 f32)
  return __builtin_amdgcn_wmma_f32_16x16x32_f16(false, a, false, b, (short)0, c,
                                                false, false);
}

__global__ __launch_bounds__(32)
void lstm_ae_kernel(const float* __restrict__ x,
                    const float* __restrict__ eWih, const float* __restrict__ eWhh,
                    const float* __restrict__ eBih, const float* __restrict__ eBhh,
                    const float* __restrict__ dWih, const float* __restrict__ dWhh,
                    const float* __restrict__ dBih, const float* __restrict__ dBhh,
                    const float* __restrict__ oW,   const float* __restrict__ oB,
                    float* __restrict__ out)
{
  // per-wave h transpose buffer: 16 rows (batch) x 16 cols (hidden), f16
  __shared__ __align__(32) _Float16 hbuf[16 * 16];

  const int lane = threadIdx.x;          // 0..31 (wave32)
  const int n    = lane & 15;            // column index within 16-wide tile
  const int lo   = (lane < 16) ? 1 : 0;
  const int boff = lo ? 0 : 8;           // C-layout row offset for this lane half
  const int ko   = lo ? 0 : 16;          // A/B-layout K offset for this lane half
  const int base = blockIdx.x * 16;      // batch tile base

  // ---------------- encoder weight fragments ----------------
  // Combined Wc[k,g]: k<8 -> Wih[g,k], 8<=k<24 -> Whh[g,k-8], else 0.
  v16h Be[4]; v8f Ce[4];
#pragma unroll
  for (int j = 0; j < 4; ++j) {
    int g = 16 * j + n;
#pragma unroll
    for (int e = 0; e < 16; ++e) {
      int k = ko + e;
      float w = 0.f;
      if (k < 8)       w = eWih[g * 8 + k];
      else if (k < 24) w = eWhh[g * 16 + (k - 8)];
      Be[j][e] = (_Float16)w;
    }
    Ce[j] = bcast8(eBih[g] + eBhh[g]);
  }

  // zero h buffer (h0 = 0)
#pragma unroll
  for (int e = 0; e < 8; ++e) hbuf[lane * 8 + e] = (_Float16)0.f;
  __syncthreads();

  v8f c = bcast8(0.f);
  v8f h = bcast8(0.f);

  const float* xptr = x + (size_t)(base + lane) * (SEQ_LEN * 8);
  v4f xa0 = {}, xa1 = {};
  if (lo) { xa0 = *(const v4f*)(xptr); xa1 = *(const v4f*)(xptr + 4); }

  // ---------------- encoder recurrence ----------------
#pragma unroll 1
  for (int t = 0; t < SEQ_LEN; ++t) {
    // A = [x_t(K0..7) | h(K8..23) | 0]
    v16h a;
    if (lo) {
#pragma unroll
      for (int e = 0; e < 4; ++e) { a[e] = (_Float16)xa0[e]; a[4 + e] = (_Float16)xa1[e]; }
      v8h hh = *(const v8h*)(&hbuf[lane * 16]);        // h[b, 0..7]
#pragma unroll
      for (int e = 0; e < 8; ++e) a[8 + e] = hh[e];
    } else {
      v8h hl = *(const v8h*)(&hbuf[(lane - 16) * 16 + 8]); // h[b, 8..15]
#pragma unroll
      for (int e = 0; e < 8; ++e) { a[e] = hl[e]; a[8 + e] = (_Float16)0.f; }
    }
    // prefetch next x_t while WMMA/elementwise run
    if (lo && (t + 1 < SEQ_LEN)) {
      xa0 = *(const v4f*)(xptr + (size_t)(t + 1) * 8);
      xa1 = *(const v4f*)(xptr + (size_t)(t + 1) * 8 + 4);
    }
    v8f gi = wmma16(a, Be[0], Ce[0]);
    v8f gf = wmma16(a, Be[1], Ce[1]);
    v8f gg = wmma16(a, Be[2], Ce[2]);
    v8f go = wmma16(a, Be[3], Ce[3]);
#pragma unroll
    for (int r = 0; r < 8; ++r) {
      float iv = fsig(gi[r]);
      float fv = fsig(gf[r]);
      float gv = ftanh(gg[r]);
      float ov = fsig(go[r]);
      c[r] = fv * c[r] + iv * gv;
      h[r] = ov * ftanh(c[r]);
    }
    __syncthreads();
#pragma unroll
    for (int r = 0; r < 8; ++r) hbuf[(boff + r) * 16 + n] = (_Float16)h[r];
    __syncthreads();
  }

  // ---------------- decoder setup ----------------
  // A_enc = h_enc in K=0..15
  v16h ae;
  {
    v8h hv = *(const v8h*)(&hbuf[(lane & 15) * 16]);
    v8h hw = *(const v8h*)(&hbuf[(lane & 15) * 16 + 8]);
#pragma unroll
    for (int e = 0; e < 8; ++e) { ae[e] = lo ? hv[e] : (_Float16)0.f; ae[8 + e] = lo ? hw[e] : (_Float16)0.f; }
  }

  v16h Bx0, Bx1, Bd0, Bd1, Bp;
#pragma unroll
  for (int e = 0; e < 16; ++e) {
    int k = ko + e;
    int g0 = n, g1 = 16 + n;
    Bx0[e] = (_Float16)((k < 16) ? dWih[g0 * 16 + k] : 0.f);
    Bx1[e] = (_Float16)((k < 16) ? dWih[g1 * 16 + k] : 0.f);
    Bd0[e] = (_Float16)((k < 8)  ? dWhh[g0 * 8 + k]  : 0.f);
    Bd1[e] = (_Float16)((k < 8)  ? dWhh[g1 * 8 + k]  : 0.f);
    Bp [e] = (_Float16)((k < 8 && n < 8) ? oW[n * 8 + k] : 0.f);
  }
  v8f CdB0 = bcast8(dBih[n] + dBhh[n]);
  v8f CdB1 = bcast8(dBih[16 + n] + dBhh[16 + n]);
  v8f Cp   = bcast8((n < 8) ? oB[n] : 0.f);

  // constant input contribution for all decoder timesteps (computed ONCE)
  v8f xgc0 = wmma16(ae, Bx0, CdB0);
  v8f xgc1 = wmma16(ae, Bx1, CdB1);

  // ---------------- decoder recurrence (+ fused output projection) ----------------
  c = bcast8(0.f);
  v16h ad;
#pragma unroll
  for (int e = 0; e < 16; ++e) ad[e] = (_Float16)0.f;  // h_dec(0) = 0
  __syncthreads();

#pragma unroll 1
  for (int t = 0; t < SEQ_LEN; ++t) {
    v8f g0 = wmma16(ad, Bd0, xgc0);   // cols: i(0..7) | f(8..15)
    v8f g1 = wmma16(ad, Bd1, xgc1);   // cols: g(0..7) | o(8..15)
#pragma unroll
    for (int r = 0; r < 8; ++r) {
      float fraw = __shfl_xor(g0[r], 8, 32);
      float oraw = __shfl_xor(g1[r], 8, 32);
      float iv = fsig(g0[r]);
      float fv = fsig(fraw);
      float gv = ftanh(g1[r]);
      float ov = fsig(oraw);
      c[r] = fv * c[r] + iv * gv;
      h[r] = ov * ftanh(c[r]);
    }
    __syncthreads();
    if (n < 8) {
#pragma unroll
      for (int r = 0; r < 8; ++r) hbuf[(boff + r) * 16 + n] = (_Float16)h[r];
    }
    __syncthreads();
    // rebuild A from new h (used for this step's projection AND next step's gates)
    if (lo) {
      v8h hl = *(const v8h*)(&hbuf[lane * 16]);
#pragma unroll
      for (int e = 0; e < 8; ++e) { ad[e] = hl[e]; ad[8 + e] = (_Float16)0.f; }
    } else {
#pragma unroll
      for (int e = 0; e < 16; ++e) ad[e] = (_Float16)0.f;
    }
    v8f p = wmma16(ad, Bp, Cp);       // out_t = h_t @ out_W.T + out_b
    if (n < 8) {
#pragma unroll
      for (int r = 0; r < 8; ++r) {
        out[((size_t)(base + boff + r) * SEQ_LEN + t) * 8 + n] = p[r];
      }
    }
  }
}

extern "C" void kernel_launch(void* const* d_in, const int* in_sizes, int n_in,
                              void* d_out, int out_size, void* d_ws, size_t ws_size,
                              hipStream_t stream) {
  (void)in_sizes; (void)n_in; (void)out_size; (void)d_ws; (void)ws_size;
  lstm_ae_kernel<<<BATCH / 16, 32, 0, stream>>>(
      (const float*)d_in[0],
      (const float*)d_in[1], (const float*)d_in[2],
      (const float*)d_in[3], (const float*)d_in[4],
      (const float*)d_in[5], (const float*)d_in[6],
      (const float*)d_in[7], (const float*)d_in[8],
      (const float*)d_in[9], (const float*)d_in[10],
      (float*)d_out);
}
